// CausalSelfAttention_64622077935919
// MI455X (gfx1250) — matrix operand order repaired and checked by reading, not compiled, not run on hardware
//
#include <hip/hip_runtime.h>
#include <math.h>

// ---------------- problem constants (match reference) ----------------
#define B_   2
#define T_   2048
#define C_   2048
#define NH   16
#define NKV  4
#define HD   128
#define G_   (NH / NKV)
#define EPS_ 1.1920928955078125e-07f

typedef __attribute__((ext_vector_type(16))) __bf16 v16bf;
typedef __attribute__((ext_vector_type(8)))  float  v8f;
typedef __attribute__((ext_vector_type(4)))  int    v4i;
typedef unsigned short u16;
typedef unsigned int   u32;

#define GL_AS  __attribute__((address_space(1)))
#define LDS_AS __attribute__((address_space(3)))

static __device__ __forceinline__ u16 f32_to_bf16(float f) {
  u32 u = __float_as_uint(f);
  u32 r = u + 0x7FFFu + ((u >> 16) & 1u);   // round-to-nearest-even
  return (u16)(r >> 16);
}

static __device__ __forceinline__ v8f wmma_bf16(v16bf a, v16bf b, v8f c) {
  // (neg_a, A, neg_b, B, c_mod, C, reuse_a, reuse_b)
  return __builtin_amdgcn_wmma_f32_16x16x32_bf16(false, a, false, b, (short)0, c,
                                                 false, false);
}

union Frag16 { v16bf v; uint4 q[2]; u16 u[16]; };

// ---- CDNA5 async global->LDS (ASYNCcnt path), with safe fallbacks ----------
static __device__ __forceinline__ void async_ld_b128(const void* g, void* l) {
#if __has_builtin(__builtin_amdgcn_global_load_async_to_lds_b128)
  __builtin_amdgcn_global_load_async_to_lds_b128(
      (GL_AS v4i*)g, (LDS_AS v4i*)l, 0, 0);
#else
  *(uint4*)l = *(const uint4*)g;
#endif
}

static __device__ __forceinline__ void wait_async0() {
#if __has_builtin(__builtin_amdgcn_s_wait_asynccnt)
  __builtin_amdgcn_s_wait_asynccnt(0);
#else
  asm volatile("s_wait_asynccnt 0x0" ::: "memory");
#endif
}

// ---------------- fp32 -> bf16 convert ----------------
__global__ void k_f32_to_bf16(const float* __restrict__ src,
                              u16* __restrict__ dst, int n) {
  int i = blockIdx.x * blockDim.x + threadIdx.x;
  int stride = gridDim.x * blockDim.x;
  for (; i < n; i += stride) dst[i] = f32_to_bf16(src[i]);
}

// ---------------- tiled bf16 WMMA GEMM: C(f32, MxN) = A(bf16, MxK) * B(bf16, KxN)
// Block tile 128x128, BK=32, 256 threads = 8 waves (2x4), wave tile 64x32.
// All dims must be multiples of 128 (K multiple of 32) -- true for this problem.
__global__ __launch_bounds__(256)
void k_gemm_bf16(const u16* __restrict__ A, const u16* __restrict__ Bm,
                 float* __restrict__ Cm, int M, int N, int K) {
  __shared__ u16 As[128 * 32];   // A tile, row-major [row][k]
  __shared__ u16 Bt[128 * 32];   // B tile, transposed [n][k]

  const int tid   = threadIdx.x;
  const int lane  = tid & 31;
  const int wave  = tid >> 5;
  const int waveM = wave >> 2;          // 0..1
  const int waveN = wave & 3;           // 0..3
  const int hf    = lane >> 4;          // half-wave id
  const int lr    = lane & 15;
  const int kbse  = hf * 8;             // 16-bit A-layout k base per half

  const int m0 = blockIdx.y * 128;
  const int n0 = blockIdx.x * 128;

  v8f zero = {};
  v8f acc[4][2];
  #pragma unroll
  for (int i = 0; i < 4; ++i)
    #pragma unroll
    for (int j = 0; j < 2; ++j) acc[i][j] = zero;

  for (int k0 = 0; k0 < K; k0 += 32) {
    // stage A tile 128x32 via async DMA (2 passes x 256 thr x 16B)
    #pragma unroll
    for (int p = 0; p < 2; ++p) {
      int idx = tid + p * 256;
      int row = idx >> 2;
      int cc  = idx & 3;
      async_ld_b128(&A[(m0 + row) * K + k0 + cc * 8], &As[row * 32 + cc * 8]);
    }
    // stage B tile 32x128, register-transposed into Bt[n][k]
    #pragma unroll
    for (int p = 0; p < 2; ++p) {
      int idx = tid + p * 256;
      int kk  = idx >> 4;                // 0..31
      int nc  = idx & 15;
      uint4 d = *(const uint4*)&Bm[(k0 + kk) * N + n0 + nc * 8];
      const u16* e = (const u16*)&d;
      #pragma unroll
      for (int j = 0; j < 8; ++j) Bt[(nc * 8 + j) * 32 + kk] = e[j];
    }
    wait_async0();          // this wave's DMA landed in LDS
    __syncthreads();        // all waves' staging (async + ds stores) visible

    // fragments per documented 16-bit A/B lane layouts
    Frag16 af[4], bf[2];
    #pragma unroll
    for (int mt = 0; mt < 4; ++mt) {
      int row = waveM * 64 + mt * 16 + lr;
      af[mt].q[0] = *(const uint4*)&As[row * 32 + kbse];        // K = kbse..+7
      af[mt].q[1] = *(const uint4*)&As[row * 32 + 16 + kbse];   // K = 16+kbse..
    }
    #pragma unroll
    for (int nt = 0; nt < 2; ++nt) {
      int col = waveN * 32 + nt * 16 + lr;
      bf[nt].q[0] = *(const uint4*)&Bt[col * 32 + hf * 8];      // K = hf*8..+7
      bf[nt].q[1] = *(const uint4*)&Bt[col * 32 + 16 + hf * 8]; // K = 16+hf*8..
    }
    #pragma unroll
    for (int mt = 0; mt < 4; ++mt)
      #pragma unroll
      for (int nt = 0; nt < 2; ++nt)
        acc[mt][nt] = wmma_bf16(af[mt].v, bf[nt].v, acc[mt][nt]);
    __syncthreads();
  }

  // epilogue: C/D layout M = r + 8*hf, N = lr
  #pragma unroll
  for (int mt = 0; mt < 4; ++mt)
    #pragma unroll
    for (int nt = 0; nt < 2; ++nt) {
      int gn = n0 + waveN * 32 + nt * 16 + lr;
      #pragma unroll
      for (int r = 0; r < 8; ++r) {
        int gm = m0 + waveM * 64 + mt * 16 + r + 8 * hf;
        Cm[gm * N + gn] = acc[mt][nt][r];
      }
    }
}

// ---------------- RoPE + RMSNorm: one wave per (b,t,head) row of 128 ----------
__global__ __launch_bounds__(256)
void k_rope_rms(const float* __restrict__ src, const float* __restrict__ cosT,
                const float* __restrict__ sinT, u16* __restrict__ dst,
                int nh, int nrows) {
  int rid  = blockIdx.x * 8 + (threadIdx.x >> 5);
  int lane = threadIdx.x & 31;
  if (rid >= nrows) return;
  int t = (rid / nh) % T_;
  const float* v = src + rid * HD;
  float y[4];
  float ss = 0.f;
  #pragma unroll
  for (int p = 0; p < 2; ++p) {
    int d   = lane + p * 32;            // 0..63
    float c = cosT[t * 64 + d];
    float s = sinT[t * 64 + d];
    float x1 = v[d];
    float x2 = v[d + 64];
    float a  =  x1 * c + x2 * s;
    float b2 = -x1 * s + x2 * c;
    y[p * 2 + 0] = a;
    y[p * 2 + 1] = b2;
    ss += a * a + b2 * b2;
  }
  #pragma unroll
  for (int off = 16; off >= 1; off >>= 1) ss += __shfl_xor(ss, off, 32);
  float r = rsqrtf(ss * (1.0f / HD) + EPS_);
  u16* o = dst + rid * HD;
  #pragma unroll
  for (int p = 0; p < 2; ++p) {
    int d = lane + p * 32;
    o[d]      = f32_to_bf16(y[p * 2 + 0] * r);
    o[d + 64] = f32_to_bf16(y[p * 2 + 1] * r);
  }
}

// ---------------- flash attention: per block 128 q rows of one (b,h) ---------
// 8 waves, 16 q rows per wave, key tiles of 32, online softmax, all WMMA bf16.
__global__ __launch_bounds__(256)
void k_attn(const u16* __restrict__ qb, const u16* __restrict__ kb,
            const u16* __restrict__ vb, u16* __restrict__ aob) {
  __shared__ u16 Qs[128 * HD];        // 32 KB q tile
  __shared__ u16 Ps[8 * 16 * 32];     // 8 KB per-wave P staging

  const int b   = blockIdx.z;
  const int h   = blockIdx.y;
  const int kv  = h / G_;
  const int qt0 = blockIdx.x * 128;

  const int tid  = threadIdx.x;
  const int lane = tid & 31;
  const int wave = tid >> 5;
  const int hf   = lane >> 4;
  const int lr   = lane & 15;
  const int kbse = hf * 8;

  // stage Q tile (rows qt0..qt0+127 of head h) via async DMA
  #pragma unroll
  for (int p = 0; p < 8; ++p) {
    int idx = tid + p * 256;
    int row = idx >> 4;
    int cc  = idx & 15;
    async_ld_b128(&qb[((b * T_ + qt0 + row) * NH + h) * HD + cc * 8],
                  &Qs[row * HD + cc * 8]);
  }
  wait_async0();
  __syncthreads();

  // Q A-fragments for the 4 d-chunks of 32 (kept resident all loop)
  Frag16 aQ[4];
  #pragma unroll
  for (int ks = 0; ks < 4; ++ks) {
    int row = wave * 16 + lr;
    aQ[ks].q[0] = *(const uint4*)&Qs[row * HD + ks * 32 + kbse];
    aQ[ks].q[1] = *(const uint4*)&Qs[row * HD + ks * 32 + 16 + kbse];
  }

  const int qrw = qt0 + wave * 16;     // this wave's first query row
  v8f zero = {};
  float m[8], l[8];
  v8f O[8];
  #pragma unroll
  for (int r = 0; r < 8; ++r) { m[r] = -3.0e38f; l[r] = 0.f; }
  #pragma unroll
  for (int d = 0; d < 8; ++d) O[d] = zero;

  const float scale = 0.088388347648318447f;   // 1/sqrt(128)
  const int nkt = (qrw + 15) / 32 + 1;         // causal trip count

  for (int kt = 0; kt < nkt; ++kt) {
    const int kk0 = kt * 32;

    // prefetch next tile's K/V lines toward the WGP while we compute
    if (kt + 1 < nkt) {
      int nk = kk0 + 32 + lane;        // 32 key rows, one per lane
      __builtin_prefetch(&kb[((b * T_ + nk) * NKV + kv) * HD], 0, 1);
      __builtin_prefetch(&vb[((b * T_ + nk) * NKV + kv) * HD], 0, 1);
    }

    // S(16x32) = Q(16x128) K^T : two N-subtiles, 4 chained d-chunk WMMAs each
    v8f S[2];
    #pragma unroll
    for (int ns = 0; ns < 2; ++ns) {
      int key = kk0 + ns * 16 + lr;
      const u16* kp = kb + ((b * T_ + key) * NKV + kv) * HD;
      v8f s = zero;
      #pragma unroll
      for (int ks = 0; ks < 4; ++ks) {
        Frag16 bK;
        bK.q[0] = *(const uint4*)&kp[ks * 32 + hf * 8];
        bK.q[1] = *(const uint4*)&kp[ks * 32 + 16 + hf * 8];
        s = wmma_bf16(aQ[ks].v, bK.v, s);
      }
      S[ns] = s;
    }
    // scale + causal mask (C/D layout: row = qrw + r + 8*hf, col key = ...+lr)
    #pragma unroll
    for (int ns = 0; ns < 2; ++ns) {
      int key = kk0 + ns * 16 + lr;
      #pragma unroll
      for (int r = 0; r < 8; ++r) {
        float sv = S[ns][r] * scale;
        int rowg = qrw + r + 8 * hf;
        S[ns][r] = (key > rowg) ? -3.0e38f : sv;
      }
    }
    // online softmax: reductions across the 16 lanes of each half-wave
    float pr[2][8];
    #pragma unroll
    for (int r = 0; r < 8; ++r) {
      float vmax = fmaxf(S[0][r], S[1][r]);
      #pragma unroll
      for (int off = 8; off >= 1; off >>= 1)
        vmax = fmaxf(vmax, __shfl_xor(vmax, off, 32));
      float newm = fmaxf(m[r], vmax);
      float corr = __expf(m[r] - newm);
      float p0 = __expf(S[0][r] - newm);
      float p1 = __expf(S[1][r] - newm);
      float rs = p0 + p1;
      #pragma unroll
      for (int off = 8; off >= 1; off >>= 1)
        rs += __shfl_xor(rs, off, 32);
      l[r] = l[r] * corr + rs;
      m[r] = newm;
      pr[0][r] = p0;
      pr[1][r] = p1;
      #pragma unroll
      for (int d = 0; d < 8; ++d) O[d][r] = O[d][r] * corr;
    }
    // restage P (16x32 bf16) through per-wave LDS into A-fragment layout
    u16* pw = &Ps[wave * 16 * 32];
    #pragma unroll
    for (int ns = 0; ns < 2; ++ns)
      #pragma unroll
      for (int r = 0; r < 8; ++r)
        pw[(r + 8 * hf) * 32 + ns * 16 + lr] = f32_to_bf16(pr[ns][r]);
    asm volatile("s_wait_dscnt 0" ::: "memory");   // DS RAW before re-read
    Frag16 aP;
    aP.q[0] = *(const uint4*)&pw[lr * 32 + hf * 16];
    aP.q[1] = *(const uint4*)&pw[lr * 32 + hf * 16 + 8];
    // O(16x128) += P(16x32) V(32x128): 8 N-subtiles of d
    const u16* vbase = vb + ((b * T_ + kk0 + hf * 16) * NKV + kv) * HD;
    #pragma unroll
    for (int d = 0; d < 8; ++d) {
      Frag16 bV;
      #pragma unroll
      for (int i = 0; i < 16; ++i)
        bV.u[i] = vbase[i * (NKV * HD) + d * 16 + lr];
      O[d] = wmma_bf16(aP.v, bV.v, O[d]);
    }
  }

  // epilogue: normalize by row sum, write bf16 attn-out [B,T,NH,HD]
  float inv[8];
  #pragma unroll
  for (int r = 0; r < 8; ++r) inv[r] = 1.0f / l[r];
  #pragma unroll
  for (int d = 0; d < 8; ++d)
    #pragma unroll
    for (int r = 0; r < 8; ++r) {
      int tg = qrw + r + 8 * hf;
      aob[((b * T_ + tg) * NH + h) * HD + d * 16 + lr] =
          f32_to_bf16(O[d][r] * inv[r]);
    }
}

// ---------------- host-side orchestration ----------------
extern "C" void kernel_launch(void* const* d_in, const int* in_sizes, int n_in,
                              void* d_out, int out_size, void* d_ws, size_t ws_size,
                              hipStream_t stream) {
  const float* x    = (const float*)d_in[0];
  const float* cosT = (const float*)d_in[1];
  const float* sinT = (const float*)d_in[2];
  const float* WQ   = (const float*)d_in[3];
  const float* WK   = (const float*)d_in[4];
  const float* WV   = (const float*)d_in[5];
  const float* WO   = (const float*)d_in[6];

  char* ws = (char*)d_ws;
  size_t off = 0;
  auto carve = [&](size_t bytes) -> void* {
    void* p = ws + off;
    off = (off + bytes + 255) & ~(size_t)255;
    return p;
  };

  const size_t R = (size_t)B_ * T_;                 // 4096 token rows
  u16*   xb  = (u16*)carve(R * C_ * 2);             // x, bf16
  u16*   wqb = (u16*)carve((size_t)C_ * NH  * HD * 2);
  u16*   wkb = (u16*)carve((size_t)C_ * NKV * HD * 2);
  u16*   wvb = (u16*)carve((size_t)C_ * NKV * HD * 2);
  u16*   wob = (u16*)carve((size_t)C_ * C_ * 2);
  float* qf  = (float*)carve(R * NH  * HD * 4);     // raw projections, f32
  float* kf  = (float*)carve(R * NKV * HD * 4);
  float* vf  = (float*)carve(R * NKV * HD * 4);
  u16*   qbn = (u16*)carve(R * NH  * HD * 2);       // rope+rms, bf16
  u16*   kbn = (u16*)carve(R * NKV * HD * 2);
  u16*   vbn = (u16*)carve(R * NKV * HD * 2);
  u16*   aob = (u16*)carve(R * C_ * 2);             // attention out, bf16

  dim3 blk(256);
  auto cvt = [&](const float* s, u16* d, int n) {
    k_f32_to_bf16<<<512, blk, 0, stream>>>(s, d, n);
  };
  cvt(x,  xb,  B_ * T_ * C_);
  cvt(WQ, wqb, C_ * NH  * HD);
  cvt(WK, wkb, C_ * NKV * HD);
  cvt(WV, wvb, C_ * NKV * HD);
  cvt(WO, wob, C_ * C_);

  // QKV projections (WMMA bf16, f32 accumulate)
  k_gemm_bf16<<<dim3((NH  * HD) / 128, (B_ * T_) / 128), blk, 0, stream>>>(
      xb, wqb, qf, B_ * T_, NH * HD, C_);
  k_gemm_bf16<<<dim3((NKV * HD) / 128, (B_ * T_) / 128), blk, 0, stream>>>(
      xb, wkb, kf, B_ * T_, NKV * HD, C_);
  k_gemm_bf16<<<dim3((NKV * HD) / 128, (B_ * T_) / 128), blk, 0, stream>>>(
      xb, wvb, vf, B_ * T_, NKV * HD, C_);

  // RoPE + RMSNorm -> bf16 q/k; plain convert for v
  k_rope_rms<<<(B_ * T_ * NH)  / 8, blk, 0, stream>>>(qf, cosT, sinT, qbn, NH,  B_ * T_ * NH);
  k_rope_rms<<<(B_ * T_ * NKV) / 8, blk, 0, stream>>>(kf, cosT, sinT, kbn, NKV, B_ * T_ * NKV);
  cvt(vf, vbn, B_ * T_ * NKV * HD);

  // flash attention (WMMA bf16 QK^T and PV)
  k_attn<<<dim3(T_ / 128, NH, B_), blk, 0, stream>>>(qbn, kbn, vbn, aob);

  // output projection into d_out (f32)
  k_gemm_bf16<<<dim3(C_ / 128, (B_ * T_) / 128), blk, 0, stream>>>(
      aob, wob, (float*)d_out, B_ * T_, C_, C_);
}
